// CapsuleMortalityHead_14491219657232
// MI455X (gfx1250) — matrix operand order repaired and checked by reading, not compiled, not run on hardware
//
#include <hip/hip_runtime.h>
#include <hip/hip_bf16.h>
#include <stdint.h>

// Problem constants (from reference)
#define B_SZ 4096
#define NR   7
#define DPC  512
#define NCLS 25
#define DMC  128
#define NK   (NCLS*DMC)   // 3200
#define ND   (NR*DPC)     // 3584
#define VOTE_PER_B (NR*NK) // 22400

typedef __attribute__((ext_vector_type(16))) __bf16 v16bf;
typedef __attribute__((ext_vector_type(8)))  float  v8f;

__device__ __forceinline__ unsigned short f2bf(float f){
  unsigned u = __float_as_uint(f);
  return (unsigned short)((u + 0x7FFFu + ((u >> 16) & 1u)) >> 16); // RNE
}
__device__ __forceinline__ float bf2f(unsigned short h){
  return __uint_as_float(((unsigned)h) << 16);
}
__device__ __forceinline__ float wred_sum(float x){
  #pragma unroll
  for (int o = 16; o > 0; o >>= 1) x += __shfl_xor(x, o, 32);
  return x;
}
__device__ __forceinline__ float wred_max(float x){
  #pragma unroll
  for (int o = 16; o > 0; o >>= 1) x = fmaxf(x, __shfl_xor(x, o, 32));
  return x;
}

// ---------------------------------------------------------------------------
// Kernel 1: f32 -> bf16 conversion for prim_pose and w; copy prim_act to out.
// ---------------------------------------------------------------------------
__global__ __launch_bounds__(256) void cvt_kernel(
    const float* __restrict__ pose, const float* __restrict__ w,
    const float* __restrict__ act,
    unsigned short* __restrict__ poseb, unsigned short* __restrict__ wb,
    float* __restrict__ actOut){
  int idx = blockIdx.x * 256 + threadIdx.x;
  if (idx < B_SZ*ND)     poseb[idx] = f2bf(pose[idx]);
  if (idx < NR*DPC*NK)   wb[idx]    = f2bf(w[idx]);
  if (idx < B_SZ*NR)     actOut[idx] = act[idx];
}

// ---------------------------------------------------------------------------
// Kernel 2: vote GEMM.  For route n: C[4096,3200] = A[4096,512] x B[512,3200]
// bf16 inputs, f32 WMMA accumulation, bf16 output (vote tensor).
// Block = 256 threads (8 wave32). Tile 256(M) x 128(N) x 32(K).
// Wave grid 4(M) x 2(N): each wave computes 64x64 = 4x4 WMMA 16x16 tiles
// (16 v_wmma per K-step against 8 b128 LDS fragment loads).
// Epilogue stages bf16 C through LDS for coalesced 128B global stores.
// ---------------------------------------------------------------------------
__global__ __launch_bounds__(256) void vote_gemm_kernel(
    const unsigned short* __restrict__ poseb,   // [B, 7, 512] bf16
    const unsigned short* __restrict__ wb,      // [7, 512, 3200] bf16
    unsigned short* __restrict__ vote){         // [B, 7, 3200] bf16
  __shared__ union {
    struct {
      unsigned A[256][16];   // [m][kpair] (dword = k even,k odd)   16 KB
      unsigned Bt[128][20];  // [col][kpair], padded 16->20          10 KB
    } t;
    unsigned short out[128][136];               // C staging (pad 128->136) 34 KB
  } sm;

  const int tid  = threadIdx.x;
  const int lane = tid & 31, wid = tid >> 5;
  const int wm = wid >> 1, wn = wid & 1;        // 4 x 2 wave grid
  const int bRow0 = blockIdx.x * 256;
  const int nCol0 = blockIdx.y * 128;
  const int rt    = blockIdx.z;

  v8f acc[4][4] = {};

  const unsigned short* Abase = poseb + (size_t)bRow0 * ND + (size_t)rt * DPC;
  const unsigned short* Bbase = wb + (size_t)rt * DPC * NK + nCol0;

  const int bkp  = tid >> 4, bseg = tid & 15;   // B loader: kpair rows x 8 cols

  for (int kk = 0; kk < DPC; kk += 32) {
    if (kk + 32 < DPC) {  // prefetch next K tile (global_prefetch_b8)
      __builtin_prefetch(Abase + (size_t)tid * ND + kk + 32, 0, 1);
      __builtin_prefetch(Bbase + (size_t)(kk + 32 + 2*bkp) * NK + bseg * 8, 0, 1);
    }
    { // A tile -> LDS: thread t loads full 32-k row for batch row t (64B)
      const uint4* src = (const uint4*)(Abase + (size_t)tid * ND + kk);
      uint4 a0 = src[0], a1 = src[1], a2 = src[2], a3 = src[3];
      uint4* dst = (uint4*)&sm.t.A[tid][0];
      dst[0] = a0; dst[1] = a1; dst[2] = a2; dst[3] = a3;
    }
    { // B tile -> LDS transposed: pack (row k, row k+1) per column into a dword
      const unsigned short* s0 = Bbase + (size_t)(kk + 2*bkp) * NK + bseg * 8;
      const unsigned short* s1 = s0 + NK;
      uint4 ra = *(const uint4*)s0;
      uint4 rb = *(const uint4*)s1;
      unsigned pa[4] = {ra.x, ra.y, ra.z, ra.w};
      unsigned pb[4] = {rb.x, rb.y, rb.z, rb.w};
      const int c0 = bseg * 8;
      #pragma unroll
      for (int j = 0; j < 4; j++){
        sm.t.Bt[c0 + 2*j  ][bkp] = (pa[j] & 0xFFFFu) | (pb[j] << 16);
        sm.t.Bt[c0 + 2*j+1][bkp] = (pa[j] >> 16)     | (pb[j] & 0xFFFF0000u);
      }
    }
    __syncthreads();

    union Frag { unsigned u[8]; uint4 q[2]; v16bf v; };
    // B frags first (reused across all i): two b128 loads each
    Frag bfr[4];
    #pragma unroll
    for (int j = 0; j < 4; j++){
      int col = wn*64 + j*16 + (lane & 15);
      const uint4* pb = (const uint4*)&sm.t.Bt[col][(lane >> 4) << 3];
      bfr[j].q[0] = pb[0]; bfr[j].q[1] = pb[1];
    }
    #pragma unroll
    for (int i = 0; i < 4; i++){
      int row = wm*64 + i*16 + (lane & 15);
      // A frag: lane<16 -> K 0..15 ; lane>=16 -> K 16..31
      Frag af;
      const uint4* pa = (const uint4*)&sm.t.A[row][(lane >> 4) << 3];
      af.q[0] = pa[0]; af.q[1] = pa[1];
      #pragma unroll
      for (int j = 0; j < 4; j++)
        acc[i][j] = __builtin_amdgcn_wmma_f32_16x16x32_bf16(
            false, af.v, false, bfr[j].v, (short)0, acc[i][j], false, false);
    }
    __syncthreads();
  }

  // Epilogue: stage bf16 C tile through LDS in two 128-row chunks,
  // then write coalesced 128B rows to the vote tensor.
  for (int h = 0; h < 2; h++){
    if ((wm >> 1) == h){
      #pragma unroll
      for (int i = 0; i < 4; i++){
        #pragma unroll
        for (int j = 0; j < 4; j++){
          #pragma unroll
          for (int r = 0; r < 8; r++){
            int lr = (wm & 1)*64 + i*16 + r + ((lane >> 4) << 3);
            int lc = wn*64 + j*16 + (lane & 15);
            sm.out[lr][lc] = f2bf(acc[i][j][r]);
          }
        }
      }
    }
    __syncthreads();
    {
      int row  = tid >> 1, half = tid & 1;
      const uint4* s = (const uint4*)&sm.out[row][half * 64];
      uint4 x0 = s[0], x1 = s[1];
      unsigned short* d = vote + (size_t)(bRow0 + h*128 + row) * VOTE_PER_B
                               + (size_t)rt * NK + nCol0 + half * 64;
      ((uint4*)d)[0] = x0; ((uint4*)d)[1] = x1;
    }
    __syncthreads();
  }
}

// ---------------------------------------------------------------------------
// Kernel 3: one routing iteration, fused per batch row.
// it==0: coef = 1/25 uniform.  it>0: coef = softmax_m(scale * vote . pose).
// Then pose[m,:] = LayerNorm(sum_n coef*act * vote[n,m,:]).
// Final call also writes routing_coef and logits.
// ---------------------------------------------------------------------------
__global__ __launch_bounds__(256) void route_kernel(
    const unsigned short* __restrict__ vote,  // [B, 7*25*128] bf16
    const float* __restrict__ act,            // [B, 7]
    float* __restrict__ pose,                 // [B, 25, 128] in/out
    const float* __restrict__ gamma,
    const float* __restrict__ beta,
    const float* __restrict__ emb,            // [25, 128]
    const float* __restrict__ bias,           // [25]
    float* __restrict__ coefOut,              // [B, 7, 25] or null
    float* __restrict__ logitsOut,            // [B, 25] or null
    int it){
  __shared__ unsigned short sv[VOTE_PER_B];   // 44.8 KB votes for this row
  __shared__ float sp[NCLS*DMC];              // 12.8 KB pose
  __shared__ float sc[NR*32];                 // coef (padded 25->32)

  const int tid = threadIdx.x, lane = tid & 31, wid = tid >> 5;
  const int b = blockIdx.x;

  const unsigned* gv = (const unsigned*)(vote + (size_t)b * VOTE_PER_B);
  unsigned* lv = (unsigned*)sv;
  for (int i = tid; i < VOTE_PER_B/2; i += 256) lv[i] = gv[i];
  if (it > 0)
    for (int i = tid; i < NCLS*DMC; i += 256) sp[i] = pose[(size_t)b*NCLS*DMC + i];
  __syncthreads();

  if (it == 0) {
    if (wid < NR && lane < NCLS) sc[wid*32 + lane] = act[b*NR + wid] * (1.0f/NCLS);
  } else {
    const float scale = 0.08838834764831845f;  // 1/sqrt(128)
    for (int p = wid; p < NR*NCLS; p += 8) {   // agreement dots
      int m = p % NCLS;
      const unsigned short* vv = &sv[p*DMC + lane*4];
      const float* pp = &sp[m*DMC + lane*4];
      float part = bf2f(vv[0])*pp[0] + bf2f(vv[1])*pp[1]
                 + bf2f(vv[2])*pp[2] + bf2f(vv[3])*pp[3];
      float s = wred_sum(part);
      if (lane == 0) sc[(p/NCLS)*32 + m] = s * scale;
    }
    __syncthreads();
    if (wid < NR) {                            // softmax over m per route
      float x = (lane < NCLS) ? sc[wid*32 + lane] : -3.4e38f;
      float mx = wred_max(x);
      float e = (lane < NCLS) ? __expf(x - mx) : 0.0f;
      float s = wred_sum(e);
      float cf = e / s;
      if (lane < NCLS) {
        if (coefOut) coefOut[(size_t)b*NR*NCLS + wid*NCLS + lane] = cf;
        sc[wid*32 + lane] = cf * act[b*NR + wid];
      }
    }
  }
  __syncthreads();

  // pose update + LayerNorm (+ logits on final iteration)
  for (int m = wid; m < NCLS; m += 8) {
    float a0=0.f, a1=0.f, a2=0.f, a3=0.f;
    int v0 = lane * 4;
    #pragma unroll
    for (int n = 0; n < NR; n++){
      float c = sc[n*32 + m];
      const unsigned short* vv = &sv[(n*NCLS + m)*DMC + v0];
      a0 += c * bf2f(vv[0]); a1 += c * bf2f(vv[1]);
      a2 += c * bf2f(vv[2]); a3 += c * bf2f(vv[3]);
    }
    float mu = wred_sum(a0 + a1 + a2 + a3) * (1.0f/DMC);
    a0 -= mu; a1 -= mu; a2 -= mu; a3 -= mu;
    float var = wred_sum(a0*a0 + a1*a1 + a2*a2 + a3*a3) * (1.0f/DMC);
    float rs = rsqrtf(var + 1e-5f);
    float y0 = a0*rs*gamma[v0]   + beta[v0];
    float y1 = a1*rs*gamma[v0+1] + beta[v0+1];
    float y2 = a2*rs*gamma[v0+2] + beta[v0+2];
    float y3 = a3*rs*gamma[v0+3] + beta[v0+3];
    float* gp = &pose[(size_t)b*NCLS*DMC + m*DMC + v0];
    gp[0]=y0; gp[1]=y1; gp[2]=y2; gp[3]=y3;
    if (logitsOut){
      const float* ee = &emb[m*DMC + v0];
      float d = wred_sum(y0*ee[0] + y1*ee[1] + y2*ee[2] + y3*ee[3]);
      if (lane == 0) logitsOut[b*NCLS + m] = d + bias[m];
    }
  }
}

// ---------------------------------------------------------------------------
extern "C" void kernel_launch(void* const* d_in, const int* in_sizes, int n_in,
                              void* d_out, int out_size, void* d_ws, size_t ws_size,
                              hipStream_t stream){
  (void)in_sizes; (void)n_in; (void)out_size; (void)ws_size;
  const float* prim_pose = (const float*)d_in[0];
  const float* prim_act  = (const float*)d_in[1];
  const float* w         = (const float*)d_in[2];
  const float* ln_gamma  = (const float*)d_in[3];
  const float* ln_beta   = (const float*)d_in[4];
  const float* embedding = (const float*)d_in[5];
  const float* bias      = (const float*)d_in[6];

  float* out     = (float*)d_out;
  float* logits  = out;                         // [4096, 25]
  float* actOut  = out + B_SZ*NCLS;             // [4096, 7]
  float* coefOut = out + B_SZ*NCLS + B_SZ*NR;   // [4096, 7, 25]

  char* ws = (char*)d_ws;
  size_t off = 0;
  unsigned short* poseb = (unsigned short*)(ws + off); off += (size_t)B_SZ*ND*2;        // 29.4 MB
  unsigned short* wb    = (unsigned short*)(ws + off); off += (size_t)NR*DPC*NK*2;      // 22.9 MB
  unsigned short* vote  = (unsigned short*)(ws + off); off += (size_t)B_SZ*VOTE_PER_B*2;// 183.5 MB
  float* pose           = (float*)(ws + off);                                           // 52.4 MB

  const int ncvt = B_SZ*ND;  // covers the largest conversion
  cvt_kernel<<<(ncvt + 255)/256, 256, 0, stream>>>(prim_pose, w, prim_act,
                                                   poseb, wb, actOut);
  dim3 g(B_SZ/256, NK/128, NR);   // (16, 25, 7)
  vote_gemm_kernel<<<g, 256, 0, stream>>>(poseb, wb, vote);

  route_kernel<<<B_SZ, 256, 0, stream>>>(vote, prim_act, pose, ln_gamma, ln_beta,
                                         embedding, bias, nullptr, nullptr, 0);
  route_kernel<<<B_SZ, 256, 0, stream>>>(vote, prim_act, pose, ln_gamma, ln_beta,
                                         embedding, bias, nullptr, nullptr, 1);
  route_kernel<<<B_SZ, 256, 0, stream>>>(vote, prim_act, pose, ln_gamma, ln_beta,
                                         embedding, bias, coefOut, logits, 2);
}
